// IEGMN_Layer_16234976379300
// MI455X (gfx1250) — compile-verified, matrix operand order checked by reading, not run on hardware
//
#include <hip/hip_runtime.h>
#include <hip/hip_bf16.h>

// ---------------- problem constants ----------------
#define N_LIG 4000
#define E_LIG 64000
#define N_REC 16000
#define E_REC 256000
#define HDIM  64
#define SLOPE 0.01f
#define X_CONN 0.25f
#define SKIP_W 0.5f

// ---------------- WMMA plumbing (gfx1250, wave32) ----------------
typedef __attribute__((ext_vector_type(16))) __bf16 v16bf;
typedef __attribute__((ext_vector_type(8)))  float  v8f;
typedef __attribute__((ext_vector_type(8)))  unsigned short us8;

union FragBF { v16bf v; us8 h2[2]; unsigned short u[16]; };

__device__ __forceinline__ unsigned short f2bf(float f) {
  union { float f; unsigned u; } v; v.f = f;
  unsigned r = v.u + 0x7FFFu + ((v.u >> 16) & 1u);   // RNE
  return (unsigned short)(r >> 16);
}

__device__ __forceinline__ float lrelu_f(float v) { return v > 0.f ? v : SLOPE * v; }

// Load a 16x32 bf16 fragment (A, or B stored as B^T [N][K]) from row-major
// ushort storage with leading dim `ld` (multiple of 8; rows 16B aligned).
// ISA 7.12.2 layout: lane holds row rowBase+(lane&15); element i maps to
// K = (i<8 ? i : i+8) + 8*(lane>=16)  -> two contiguous 8-halfword loads.
__device__ __forceinline__ v16bf frag16(const unsigned short* base, int rowBase, int ld, int k0) {
  int lane = threadIdx.x & 31;
  const unsigned short* p =
      base + (size_t)(rowBase + (lane & 15)) * ld + k0 + ((lane >> 4) << 3);
  FragBF f;
  f.h2[0] = *(const us8*)(p);
  f.h2[1] = *(const us8*)(p + 16);
  return f.v;
}

__device__ __forceinline__ v8f wmma_bf16(v16bf a, v16bf b, v8f c) {
  return __builtin_amdgcn_wmma_f32_16x16x32_bf16(false, a, false, b, (short)0, c, false, false);
}

// Wave-relative LDS byte address of a __shared__ object (generic-pointer low
// 32 bits are the LDS address the hardware uses; ISA 10.2 aperture rules).
__device__ __forceinline__ unsigned lds_addr_of(const void* p) {
  return (unsigned)(uintptr_t)p;
}

// Async 16B global->LDS copy, tracked by ASYNCcnt (ISA 10 / 15.18.3 op 98).
__device__ __forceinline__ void async_ld16(unsigned ldsByteOff, const float* base, int byteOff) {
  asm volatile("global_load_async_to_lds_b128 %0, %1, %2"
               :: "v"(ldsByteOff), "v"(byteOff), "s"(base)
               : "memory");
}

// ---------------- small utility kernels ----------------
__global__ void conv_wT_kernel(const float* __restrict__ W, unsigned short* __restrict__ Wt,
                               int K, int N, int Kpad) {
  int idx = blockIdx.x * 256 + threadIdx.x;
  if (idx >= N * Kpad) return;
  int n = idx / Kpad, k = idx - n * Kpad;
  Wt[(size_t)n * Kpad + k] = (k < K) ? f2bf(W[(size_t)k * N + n]) : (unsigned short)0;
}

__global__ void zero_kernel(float* __restrict__ p, int n) {
  int i = blockIdx.x * 256 + threadIdx.x;
  if (i < n) p[i] = 0.f;
}

// ---------------- edge branch: msg MLP + coef MLP + scatter ----------------
__global__ void __launch_bounds__(256) edge_kernel(
    const float* __restrict__ coords, const float* __restrict__ h,
    const float* __restrict__ efeat, const int* __restrict__ src,
    const int* __restrict__ dst, int EF, float invDistScale,
    const unsigned short* __restrict__ W1t, const float* __restrict__ b1,   // [64][192]
    const unsigned short* __restrict__ W2t, const float* __restrict__ b2,   // [64][64]
    const unsigned short* __restrict__ cW1t, const float* __restrict__ cb1, // [64][64]
    const float* __restrict__ cW2, const float* __restrict__ cb2,           // [64],[1]
    float* __restrict__ xsum, float* __restrict__ aggr, float* __restrict__ cnt) {
  __shared__ __align__(16) unsigned short featA[32 * 192];
  __shared__ __align__(16) unsigned short msgB[32 * 64];
  __shared__ float msgF[32 * 64];
  __shared__ float sqS[32], coefS[32], cW2S[64];
  __shared__ float xrelS[32][3];
  __shared__ int srcS[32], dstS[32];

  int tid = threadIdx.x, lane = tid & 31, wave = tid >> 5;
  int e0 = blockIdx.x * 32;
  if (tid < 64) cW2S[tid] = cW2[tid];
  if (tid < 32) {
    int e = e0 + tid;
    int s = src[e], d = dst[e];
    srcS[tid] = s; dstS[tid] = d;
    float sq = 0.f;
    #pragma unroll
    for (int c = 0; c < 3; ++c) {
      float xr = coords[(size_t)s * 3 + c] - coords[(size_t)d * 3 + c];
      xrelS[tid][c] = xr;
      sq += xr * xr;
    }
    sqS[tid] = sq * invDistScale;
    coefS[tid] = cb2[0];
  }
  __syncthreads();

  // features: [h_src(64) | h_dst(64) | efeat(EF) | rbf(15) | 0-pad -> 192]
  for (int idx = tid; idx < 32 * 192; idx += 256) {
    int e = idx / 192, c = idx - e * 192;
    float v = 0.f;
    if (c < 64)               v = h[(size_t)srcS[e] * HDIM + c];
    else if (c < 128)         v = h[(size_t)dstS[e] * HDIM + (c - 64)];
    else if (c < 128 + EF)    v = efeat[(size_t)(e0 + e) * EF + (c - 128)];
    else if (c < 128 + EF + 15) {
      int j = c - 128 - EF;
      v = __expf(-sqS[e] / __powf(1.5f, (float)j));
    }
    featA[idx] = f2bf(v);
  }
  __syncthreads();

  int m0 = (wave >> 2) << 4, n0 = (wave & 3) << 4;
  int col = n0 + (lane & 15);
  int rhi = (lane >> 4) << 3;

  // GEMM1: [32x192] @ W1 -> lrelu -> bf16
  v8f c1 = {};
  for (int kk = 0; kk < 192; kk += 32)
    c1 = wmma_bf16(frag16(featA, m0, 192, kk), frag16(W1t, n0, 192, kk), c1);
  #pragma unroll
  for (int r = 0; r < 8; ++r) {
    int row = m0 + r + rhi;
    msgB[row * 64 + col] = f2bf(lrelu_f(c1[r] + b1[col]));
  }
  __syncthreads();

  // GEMM2: msg = hidden @ W2 + b2
  v8f c2 = {};
  for (int kk = 0; kk < 64; kk += 32)
    c2 = wmma_bf16(frag16(msgB, m0, 64, kk), frag16(W2t, n0, 64, kk), c2);
  __syncthreads();  // all msgB reads done before reuse
  #pragma unroll
  for (int r = 0; r < 8; ++r) {
    int row = m0 + r + rhi;
    float v = c2[r] + b2[col];
    msgF[row * 64 + col] = v;
    msgB[row * 64 + col] = f2bf(v);
  }
  __syncthreads();

  // GEMM3: coef head: lrelu(msg @ cW1 + cb1) . cW2 + cb2 (LDS-atomic row reduce)
  v8f c3 = {};
  for (int kk = 0; kk < 64; kk += 32)
    c3 = wmma_bf16(frag16(msgB, m0, 64, kk), frag16(cW1t, n0, 64, kk), c3);
  #pragma unroll
  for (int r = 0; r < 8; ++r) {
    int row = m0 + r + rhi;
    atomicAdd(&coefS[row], lrelu_f(c3[r] + cb1[col]) * cW2S[col]);
  }
  __syncthreads();

  // scatter: segment sums (means finalized later); accumulators are L2-resident
  for (int idx = tid; idx < 32 * 64; idx += 256) {
    int e = idx >> 6, c = idx & 63;
    atomicAdd(&aggr[(size_t)dstS[e] * HDIM + c], msgF[idx]);
  }
  if (tid < 96) {
    int e = tid / 3, c = tid - e * 3;
    atomicAdd(&xsum[(size_t)dstS[e] * 3 + c], xrelS[e][c] * coefS[e]);
  }
  if (tid < 32) atomicAdd(&cnt[dstS[tid]], 1.0f);
}

// ---------------- finalize: seg-mean + coordinate evolution ----------------
__global__ void finalize_kernel(const float* __restrict__ xsum, float* __restrict__ aggr,
                                const float* __restrict__ cnt,
                                const float* __restrict__ coords,
                                const float* __restrict__ origCoords, int nNodes,
                                float* __restrict__ outX) {
  int idx = blockIdx.x * 256 + threadIdx.x;
  if (idx >= nNodes * HDIM) return;
  int i = idx >> 6, c = idx & 63;
  float denom = fmaxf(cnt[i], 1.0f);
  aggr[idx] = aggr[idx] / denom;  // in-place mean
  if (c < 3) {
    float xu = xsum[i * 3 + c] / denom;
    outX[i * 3 + c] = X_CONN * origCoords[i * 3 + c] + (1.0f - X_CONN) * coords[i * 3 + c] + xu;
  }
}

// ---------------- attention projection: out = act(h @ W) as bf16 ----------------
__global__ void __launch_bounds__(256) proj_kernel(const float* __restrict__ h,
                                                   const unsigned short* __restrict__ Wt,
                                                   int applyLrelu,
                                                   unsigned short* __restrict__ out) {
  __shared__ __align__(16) unsigned short hS[32 * 64];
  int tid = threadIdx.x, lane = tid & 31, wave = tid >> 5;
  int r0 = blockIdx.x * 32;
  for (int idx = tid; idx < 32 * 64; idx += 256) hS[idx] = f2bf(h[(size_t)r0 * HDIM + idx]);
  __syncthreads();
  int m0 = (wave >> 2) << 4, n0 = (wave & 3) << 4;
  int col = n0 + (lane & 15), rhi = (lane >> 4) << 3;
  v8f c = {};
  for (int kk = 0; kk < 64; kk += 32)
    c = wmma_bf16(frag16(hS, m0, 64, kk), frag16(Wt, n0, 64, kk), c);
  #pragma unroll
  for (int r = 0; r < 8; ++r) {
    int row = m0 + r + rhi;
    float v = c[r];
    if (applyLrelu) v = lrelu_f(v);
    out[(size_t)(r0 + row) * HDIM + col] = f2bf(v);
  }
}

// ---------------- masked flash cross-attention ----------------
// 32 query rows / block; keys streamed in blocks of 64.
// The mask tile (the only unique-per-block HBM stream) is double-buffered in
// LDS via GLOBAL_LOAD_ASYNC_TO_LDS_B128 (ASYNCcnt): issue next tile, compute
// on current, s_wait_asynccnt 2 gates the older tile (async loads complete in
// order). Transposed direction stages 16B segments contiguous along qi, so
// both directions stream the mask at full cacheline utilization.
__global__ void __launch_bounds__(256) attn_kernel(
    const unsigned short* __restrict__ qB, const unsigned short* __restrict__ kB,
    const unsigned short* __restrict__ vB, const float* __restrict__ maskP, int maskLd,
    int maskTransposed, float* __restrict__ out, int Nk) {
  __shared__ __align__(16) unsigned short qS[32 * 64];
  __shared__ __align__(16) unsigned short kS[64 * 64];  // B^T of K^T == K rows [j][d]
  __shared__ __align__(16) unsigned short vS[64 * 64];  // B^T of V  == [d][j]
  __shared__ __align__(16) unsigned short pS[32 * 64];
  __shared__ __align__(16) float sS[32 * 64];
  __shared__ __align__(16) float maskS[2][32 * 64];     // async-staged mask tiles
  __shared__ float mRun[32], lRun[32], alphaS[32];

  int tid = threadIdx.x, lane = tid & 31, wave = tid >> 5;
  int m0 = (wave >> 2) << 4, n0 = (wave & 3) << 4;
  int col = n0 + (lane & 15), rhi = (lane >> 4) << 3;
  int q0 = blockIdx.x * 32;

  // Issue one 32x64-float mask tile as 512 x 16B async segments (2/thread).
  // LDS layout: !transposed -> [q][j] (ld 64); transposed -> [j][q] (ld 32).
  // In both cases segment s occupies LDS bytes [s*16, s*16+16).
  // Addresses are clamped (never read when kj >= Nk) so every wave issues a
  // uniform number of async ops -> fixed s_wait_asynccnt thresholds are valid.
  auto issue_mask = [&](int buf, int jbase) {
    unsigned ldsBase = lds_addr_of(&maskS[buf][0]);
    #pragma unroll
    for (int t = 0; t < 2; ++t) {
      int s = tid + t * 256;
      int byteOff;
      if (!maskTransposed) {
        int q = s >> 4, j4 = (s & 15) << 2;
        int jj = jbase + j4;
        if (jj > Nk - 4) jj = Nk - 4;
        byteOff = ((q0 + q) * maskLd + jj) << 2;
      } else {
        int j = s >> 3, q4 = (s & 7) << 2;
        int jj = jbase + j;
        if (jj >= Nk) jj = Nk - 1;
        byteOff = (jj * maskLd + q0 + q4) << 2;
      }
      async_ld16(ldsBase + (unsigned)(s << 4), maskP, byteOff);
    }
  };

  for (int idx = tid; idx < 32 * 64; idx += 256) qS[idx] = qB[(size_t)q0 * HDIM + idx];
  if (tid < 32) { mRun[tid] = -1e30f; lRun[tid] = 0.f; }
  issue_mask(0, 0);

  int nIter = (Nk + 63) >> 6;
  v8f acc = {};
  for (int it = 0; it < nIter; ++it) {
    int j0 = it << 6;
    int cur = it & 1;

    // stage K/V blocks (L2-resident, reused by every block) while mask asyncs fly
    for (int idx = tid; idx < 64 * 64; idx += 256) {
      int j = idx >> 6, d = idx & 63;
      int jj = j0 + j;
      unsigned short kv = 0, vv = 0;
      if (jj < Nk) { kv = kB[(size_t)jj * HDIM + d]; vv = vB[(size_t)jj * HDIM + d]; }
      kS[idx] = kv;
      vS[d * 64 + j] = vv;
    }
    // prefetch next mask tile, then gate on the current one
    if (it + 1 < nIter) {
      issue_mask(cur ^ 1, j0 + 64);
      asm volatile("s_wait_asynccnt 0x2" ::: "memory");
    } else {
      asm volatile("s_wait_asynccnt 0x0" ::: "memory");
    }
    __syncthreads();

    // S = Q @ K^T
    v8f s = {};
    for (int kk = 0; kk < 64; kk += 32)
      s = wmma_bf16(frag16(qS, m0, 64, kk), frag16(kS, n0, 64, kk), s);

    // mask from LDS tile + stash logits
    int kj = j0 + col;
    #pragma unroll
    for (int r = 0; r < 8; ++r) {
      int row = m0 + r + rhi;
      float val;
      if (kj < Nk) {
        float mv = maskTransposed ? maskS[cur][col * 32 + row] : maskS[cur][row * 64 + col];
        val = mv * s[r] - 1000.0f * (1.0f - mv);
      } else {
        val = -1e30f;  // padded keys contribute exp(.)==0
      }
      sS[row * 64 + col] = val;
    }
    __syncthreads();

    // online softmax update per query row
    if (tid < 32) {
      float bm = mRun[tid];
      for (int j = 0; j < 64; ++j) bm = fmaxf(bm, sS[tid * 64 + j]);
      float al = __expf(mRun[tid] - bm);
      float sum = 0.f;
      for (int j = 0; j < 64; ++j) {
        float pv = __expf(sS[tid * 64 + j] - bm);
        sum += pv;
        pS[tid * 64 + j] = f2bf(pv);
      }
      mRun[tid] = bm;
      lRun[tid] = lRun[tid] * al + sum;
      alphaS[tid] = al;
    }
    __syncthreads();

    // rescale running O and accumulate P @ V
    #pragma unroll
    for (int r = 0; r < 8; ++r) acc[r] = acc[r] * alphaS[m0 + r + rhi];
    for (int kk = 0; kk < 64; kk += 32)
      acc = wmma_bf16(frag16(pS, m0, 64, kk), frag16(vS, n0, 64, kk), acc);
    __syncthreads();
  }

  #pragma unroll
  for (int r = 0; r < 8; ++r) {
    int row = m0 + r + rhi;
    out[(size_t)(q0 + row) * HDIM + col] = acc[r] / fmaxf(lRun[row], 1e-30f);
  }
}

// ---------------- node MLP: concat(256) -> 64 -> 64, skip ----------------
__global__ void __launch_bounds__(256) node_kernel(
    const float* __restrict__ h, const float* __restrict__ aggrMean,
    const float* __restrict__ att, const float* __restrict__ origH,
    const unsigned short* __restrict__ W1t, const float* __restrict__ b1,  // [64][256]
    const unsigned short* __restrict__ W2t, const float* __restrict__ b2,  // [64][64]
    float* __restrict__ outH) {
  __shared__ __align__(16) unsigned short featA[32 * 256];
  __shared__ __align__(16) unsigned short hid[32 * 64];
  int tid = threadIdx.x, lane = tid & 31, wave = tid >> 5;
  int r0 = blockIdx.x * 32;
  for (int idx = tid; idx < 32 * 256; idx += 256) {
    int row = idx >> 8, c = idx & 255;
    size_t gi = (size_t)(r0 + row) * HDIM;
    float v;
    if (c < 64)       v = h[gi + c];
    else if (c < 128) v = aggrMean[gi + (c - 64)];
    else if (c < 192) v = att[gi + (c - 128)];
    else              v = origH[gi + (c - 192)];
    featA[idx] = f2bf(v);
  }
  __syncthreads();
  int m0 = (wave >> 2) << 4, n0 = (wave & 3) << 4;
  int col = n0 + (lane & 15), rhi = (lane >> 4) << 3;
  v8f c1 = {};
  for (int kk = 0; kk < 256; kk += 32)
    c1 = wmma_bf16(frag16(featA, m0, 256, kk), frag16(W1t, n0, 256, kk), c1);
  #pragma unroll
  for (int r = 0; r < 8; ++r) {
    int row = m0 + r + rhi;
    hid[row * 64 + col] = f2bf(lrelu_f(c1[r] + b1[col]));
  }
  __syncthreads();
  v8f c2 = {};
  for (int kk = 0; kk < 64; kk += 32)
    c2 = wmma_bf16(frag16(hid, m0, 64, kk), frag16(W2t, n0, 64, kk), c2);
  #pragma unroll
  for (int r = 0; r < 8; ++r) {
    int row = m0 + r + rhi;
    size_t gi = (size_t)(r0 + row) * HDIM;
    outH[gi + col] = SKIP_W * (c2[r] + b2[col]) + (1.0f - SKIP_W) * h[gi + col];
  }
}

// ---------------- host launcher ----------------
extern "C" void kernel_launch(void* const* d_in, const int* in_sizes, int n_in,
                              void* d_out, int out_size, void* d_ws, size_t ws_size,
                              hipStream_t stream) {
  (void)in_sizes; (void)n_in; (void)out_size; (void)ws_size;
  const float* coords_lig      = (const float*)d_in[0];
  const float* h_lig           = (const float*)d_in[1];
  const float* orig_h_lig      = (const float*)d_in[2];
  const float* orig_coords_lig = (const float*)d_in[3];
  const float* lig_edge_feat   = (const float*)d_in[4];
  const float* coords_rec      = (const float*)d_in[5];
  const float* h_rec           = (const float*)d_in[6];
  const float* orig_h_rec      = (const float*)d_in[7];
  const float* orig_coords_rec = (const float*)d_in[8];
  const float* rec_edge_feat   = (const float*)d_in[9];
  const float* mask            = (const float*)d_in[10];
  const float* lig_e_W1 = (const float*)d_in[11]; const float* lig_e_b1 = (const float*)d_in[12];
  const float* lig_e_W2 = (const float*)d_in[13]; const float* lig_e_b2 = (const float*)d_in[14];
  const float* rec_e_W1 = (const float*)d_in[15]; const float* rec_e_b1 = (const float*)d_in[16];
  const float* rec_e_W2 = (const float*)d_in[17]; const float* rec_e_b2 = (const float*)d_in[18];
  const float* lig_c_W1 = (const float*)d_in[19]; const float* lig_c_b1 = (const float*)d_in[20];
  const float* lig_c_W2 = (const float*)d_in[21]; const float* lig_c_b2 = (const float*)d_in[22];
  const float* rec_c_W1 = (const float*)d_in[23]; const float* rec_c_b1 = (const float*)d_in[24];
  const float* rec_c_W2 = (const float*)d_in[25]; const float* rec_c_b2 = (const float*)d_in[26];
  const float* WQ_lig = (const float*)d_in[27];
  const float* WK_lig = (const float*)d_in[28];
  const float* WV_lig = (const float*)d_in[29];
  const float* WQ     = (const float*)d_in[30];
  const float* WK     = (const float*)d_in[31];
  const float* WV     = (const float*)d_in[32];
  const float* lig_n_W1 = (const float*)d_in[33]; const float* lig_n_b1 = (const float*)d_in[34];
  const float* lig_n_W2 = (const float*)d_in[35]; const float* lig_n_b2 = (const float*)d_in[36];
  const float* rec_n_W1 = (const float*)d_in[37]; const float* rec_n_b1 = (const float*)d_in[38];
  const float* rec_n_W2 = (const float*)d_in[39]; const float* rec_n_b2 = (const float*)d_in[40];
  const int* lig_src = (const int*)d_in[41]; const int* lig_dst = (const int*)d_in[42];
  const int* rec_src = (const int*)d_in[43]; const int* rec_dst = (const int*)d_in[44];

  float* out = (float*)d_out;
  float* outXLig = out;                       // [4000,3]
  float* outHLig = out + N_LIG * 3;           // [4000,64]
  float* outXRec = outHLig + N_LIG * HDIM;    // [16000,3]
  float* outHRec = outXRec + N_REC * 3;       // [16000,64]

  // ---- workspace carve ----
  char* p = (char*)d_ws;
  auto carveB = [&](size_t bytes) -> void* {
    void* r = (void*)p;
    p += (bytes + 255) & ~(size_t)255;
    return r;
  };
  unsigned short* ligE1t = (unsigned short*)carveB(64 * 192 * 2);
  unsigned short* recE1t = (unsigned short*)carveB(64 * 192 * 2);
  unsigned short* ligE2t = (unsigned short*)carveB(64 * 64 * 2);
  unsigned short* recE2t = (unsigned short*)carveB(64 * 64 * 2);
  unsigned short* ligC1t = (unsigned short*)carveB(64 * 64 * 2);
  unsigned short* recC1t = (unsigned short*)carveB(64 * 64 * 2);
  unsigned short* WQligT = (unsigned short*)carveB(64 * 64 * 2);
  unsigned short* WKligT = (unsigned short*)carveB(64 * 64 * 2);
  unsigned short* WVligT = (unsigned short*)carveB(64 * 64 * 2);
  unsigned short* WQt    = (unsigned short*)carveB(64 * 64 * 2);
  unsigned short* WKt    = (unsigned short*)carveB(64 * 64 * 2);
  unsigned short* WVt    = (unsigned short*)carveB(64 * 64 * 2);
  unsigned short* ligN1t = (unsigned short*)carveB(64 * 256 * 2);
  unsigned short* recN1t = (unsigned short*)carveB(64 * 256 * 2);
  unsigned short* ligN2t = (unsigned short*)carveB(64 * 64 * 2);
  unsigned short* recN2t = (unsigned short*)carveB(64 * 64 * 2);

  const int ACC_TOTAL = N_LIG * 3 + N_LIG + N_LIG * HDIM + N_REC * 3 + N_REC + N_REC * HDIM;
  float* accum    = (float*)carveB((size_t)ACC_TOTAL * 4);
  float* xsumLig  = accum;
  float* cntLig   = xsumLig + N_LIG * 3;
  float* aggrLig  = cntLig + N_LIG;
  float* xsumRec  = aggrLig + N_LIG * HDIM;
  float* cntRec   = xsumRec + N_REC * 3;
  float* aggrRec  = cntRec + N_REC;

  unsigned short* qLig = (unsigned short*)carveB((size_t)N_LIG * HDIM * 2);
  unsigned short* kLig = (unsigned short*)carveB((size_t)N_LIG * HDIM * 2);
  unsigned short* vLig = (unsigned short*)carveB((size_t)N_LIG * HDIM * 2);
  unsigned short* qRec = (unsigned short*)carveB((size_t)N_REC * HDIM * 2);
  unsigned short* kRec = (unsigned short*)carveB((size_t)N_REC * HDIM * 2);
  unsigned short* vRec = (unsigned short*)carveB((size_t)N_REC * HDIM * 2);
  float* attLig = (float*)carveB((size_t)N_LIG * HDIM * 4);
  float* attRec = (float*)carveB((size_t)N_REC * HDIM * 4);

  // ---- 1) weight convert/transpose to bf16 ----
  struct CV { const float* W; unsigned short* T; int K, Kpad; };
  const CV cvs[] = {
    {lig_e_W1, ligE1t, 158, 192}, {rec_e_W1, recE1t, 170, 192},
    {lig_e_W2, ligE2t, 64, 64},   {rec_e_W2, recE2t, 64, 64},
    {lig_c_W1, ligC1t, 64, 64},   {rec_c_W1, recC1t, 64, 64},
    {WQ_lig, WQligT, 64, 64},     {WK_lig, WKligT, 64, 64},
    {WV_lig, WVligT, 64, 64},     {WQ, WQt, 64, 64},
    {WK, WKt, 64, 64},            {WV, WVt, 64, 64},
    {lig_n_W1, ligN1t, 256, 256}, {rec_n_W1, recN1t, 256, 256},
    {lig_n_W2, ligN2t, 64, 64},   {rec_n_W2, recN2t, 64, 64},
  };
  for (const CV& c : cvs) {
    int n = 64 * c.Kpad;
    conv_wT_kernel<<<(n + 255) / 256, 256, 0, stream>>>(c.W, c.T, c.K, 64, c.Kpad);
  }

  // ---- 2) zero accumulators (atomics accumulate every call) ----
  zero_kernel<<<(ACC_TOTAL + 255) / 256, 256, 0, stream>>>(accum, ACC_TOTAL);

  // ---- 3) edge branches ----
  edge_kernel<<<E_LIG / 32, 256, 0, stream>>>(
      coords_lig, h_lig, lig_edge_feat, lig_src, lig_dst, 15, 1.0f,
      ligE1t, lig_e_b1, ligE2t, lig_e_b2, ligC1t, lig_c_b1, lig_c_W2, lig_c_b2,
      xsumLig, aggrLig, cntLig);
  edge_kernel<<<E_REC / 32, 256, 0, stream>>>(
      coords_rec, h_rec, rec_edge_feat, rec_src, rec_dst, 27, 1.0f,
      recE1t, rec_e_b1, recE2t, rec_e_b2, recC1t, rec_c_b1, rec_c_W2, rec_c_b2,
      xsumRec, aggrRec, cntRec);

  // ---- 4) finalize seg-means + coordinate evolution -> d_out ----
  finalize_kernel<<<(N_LIG * HDIM + 255) / 256, 256, 0, stream>>>(
      xsumLig, aggrLig, cntLig, coords_lig, orig_coords_lig, N_LIG, outXLig);
  finalize_kernel<<<(N_REC * HDIM + 255) / 256, 256, 0, stream>>>(
      xsumRec, aggrRec, cntRec, coords_rec, orig_coords_rec, N_REC, outXRec);

  // ---- 5) attention projections (bf16) ----
  proj_kernel<<<N_LIG / 32, 256, 0, stream>>>(h_lig, WQligT, 1, qLig);
  proj_kernel<<<N_LIG / 32, 256, 0, stream>>>(h_lig, WKligT, 1, kLig);
  proj_kernel<<<N_LIG / 32, 256, 0, stream>>>(h_lig, WVligT, 0, vLig);
  proj_kernel<<<N_REC / 32, 256, 0, stream>>>(h_rec, WQt, 1, qRec);
  proj_kernel<<<N_REC / 32, 256, 0, stream>>>(h_rec, WKt, 1, kRec);
  proj_kernel<<<N_REC / 32, 256, 0, stream>>>(h_rec, WVt, 0, vRec);

  // ---- 6) masked flash cross-attention (mask streamed once each way) ----
  attn_kernel<<<N_LIG / 32, 256, 0, stream>>>(qLig, kRec, vRec, mask, N_REC, 0, attLig, N_REC);
  attn_kernel<<<N_REC / 32, 256, 0, stream>>>(qRec, kLig, vLig, mask, N_REC, 1, attRec, N_LIG);

  // ---- 7) node MLPs with skip -> d_out ----
  node_kernel<<<N_LIG / 32, 256, 0, stream>>>(h_lig, aggrLig, attLig, orig_h_lig,
                                              ligN1t, lig_n_b1, ligN2t, lig_n_b2, outHLig);
  node_kernel<<<N_REC / 32, 256, 0, stream>>>(h_rec, aggrRec, attRec, orig_h_rec,
                                              recN1t, rec_n_b1, recN2t, rec_n_b2, outHRec);
}
